// LSTMLayer_18700287607612
// MI455X (gfx1250) — matrix operand order repaired and checked by reading, not compiled, not run on hardware
//
#include <hip/hip_runtime.h>
#include <hip/hip_bf16.h>

// LSTM: B=128, T=256, D=512, H=1024, C=1000, forget_bias=1.0
// bf16 WMMA (v_wmma_f32_16x16x32_bf16), fp32 accumulation.
//  - x-part of the gate GEMM hoisted out of the recurrence into one big
//    parallel WMMA GEMM (z_x, fp32, accumulator layout) when ws_size allows;
//    the recurrent chain then runs only the 32 h-dependent K-iters per step.
//  - Per-step kernel stages Wk B-tiles into LDS with
//    global_load_async_to_lds_b128, TRIPLE buffered, prefetch distance 2,
//    synced with partial s_wait_asynccnt 0x1 + workgroup barrier.
//  - A-tiles double-buffered in registers; LDS chunk XOR-swizzle avoids
//    bank conflicts on the 64-bank LDS.

typedef __attribute__((ext_vector_type(16))) __bf16 v16bf;
typedef __attribute__((ext_vector_type(8)))  float  v8f;

#define LSTM_B 128
#define LSTM_T 256
#define LSTM_D 512
#define LSTM_H 1024
#define LSTM_C 1000
#define LSTM_CP 1008          // C padded to multiple of 16

// ---- workspace layout (bytes) ----
#define WS_XP   0                       // bf16 x packed: 128*256*512*2   = 33554432
#define WS_WKP  33554432u               // bf16 Wk packed: 1536*4096*2    = 12582912
#define WS_WP   46137344u               // bf16 w packed: 1024*1008*2     = 2064384
#define WS_HP0  48201728u               // bf16 h buf 0: 128*1024*2       = 262144
#define WS_HP1  48463872u               // bf16 h buf 1: 128*1024*2       = 262144
#define WS_C    48726016u               // fp32 c state: 128*1024*4       = 524288
#define WS_ZX   49250304u               // fp32 z_x: 128*256*4096*4       = 536870912
#define WS_ZX_END (49250304ull + 536870912ull)

static __device__ __forceinline__ float sigm(float x) {
    return 1.0f / (1.0f + __expf(-x));
}

#define WMMA_BF16(A, Bm, Cm) \
    __builtin_amdgcn_wmma_f32_16x16x32_bf16(false, (A), false, (Bm), (short)0, (Cm), false, false)

// ---------------- packing kernels ----------------

// A-tile layout (16x32 bf16 tile, 512 elems): elem (m,k) ->
//   lane = m + 16*((k%16)>=8), halfidx = (k%8) + 8*(k/16), flat = lane*16+halfidx
// x packed as [t][rowtile(8)][ktile(16)][512]
__global__ __launch_bounds__(256) void pack_x_kernel(const float* __restrict__ x,
                                                     __bf16* __restrict__ xp) {
    size_t idx = (size_t)blockIdx.x * 256 + threadIdx.x;      // over 128*256*512
    int d = idx & 511;
    int t = (int)((idx >> 9) & 255);
    int b = (int)(idx >> 17);
    float v = x[idx];
    int rt = b >> 4, m = b & 15;
    int kt = d >> 5, kk = d & 31;
    int lane = m + (((kk & 15) >= 8) ? 16 : 0);
    int hidx = (kk & 7) + 8 * (kk >> 4);
    xp[(((size_t)t * 8 + rt) * 16 + kt) * 512 + lane * 16 + hidx] = (__bf16)v;
}

// B-tile layout (32x16 bf16 tile, 512 elems): elem (k,n) ->
//   lane = n + 16*((k%16)>=8), halfidx = (k%8) + 8*(k/16); 32B lane-chunks XOR-swizzled
//   (chunk c at c ^ (c>>3)) for bank-conflict-free LDS reads.
// Wk packed as [ktile(48)][ntile(256)][512]
__global__ __launch_bounds__(256) void pack_wk_kernel(const float* __restrict__ Wk,
                                                      __bf16* __restrict__ Wkp) {
    size_t idx = (size_t)blockIdx.x * 256 + threadIdx.x;      // over 1536*4096
    int n = idx & 4095;
    int k = (int)(idx >> 12);
    float v = Wk[idx];
    int kt = k >> 5, kk = k & 31;
    int nt = n >> 4, nn = n & 15;
    int lane = nn + (((kk & 15) >= 8) ? 16 : 0);
    int sw = lane ^ (lane >> 3);
    int hidx = (kk & 7) + 8 * (kk >> 4);
    Wkp[((size_t)kt * 256 + nt) * 512 + sw * 16 + hidx] = (__bf16)v;
}

// w [1024,1000] -> padded [32 ktiles][63 ntiles][512]; same swizzle.
__global__ __launch_bounds__(256) void pack_w_kernel(const float* __restrict__ w,
                                                     __bf16* __restrict__ wp) {
    size_t idx = (size_t)blockIdx.x * 256 + threadIdx.x;      // over 1024*1008
    if (idx >= (size_t)LSTM_H * LSTM_CP) return;
    int k = (int)(idx / LSTM_CP);
    int n = (int)(idx % LSTM_CP);
    float v = (n < LSTM_C) ? w[(size_t)k * LSTM_C + n] : 0.0f;
    int kt = k >> 5, kk = k & 31;
    int nt = n >> 4, nn = n & 15;
    int lane = nn + (((kk & 15) >= 8) ? 16 : 0);
    int sw = lane ^ (lane >> 3);
    int hidx = (kk & 7) + 8 * (kk >> 4);
    wp[((size_t)kt * 63 + nt) * 512 + sw * 16 + hidx] = (__bf16)v;
}

// ---------------- z_x precompute: z_x = x @ Wx (parallel over all t) -------
// One 16x16 tile per wave. blockIdx.x = t*256 + nt, wave -> row tile.
// z_x stored fp32 in accumulator layout: [t][rt][ct][gate][lane][8].
__global__ __launch_bounds__(256) void zx_kernel(const __bf16* __restrict__ xp,
                                                 const __bf16* __restrict__ Wkp,
                                                 float* __restrict__ zx) {
    const int lane = threadIdx.x & 31;
    const int rt   = threadIdx.x >> 5;
    const int bx   = blockIdx.x;
    const int t    = bx >> 8;
    const int nt   = bx & 255;           // 0..255 = gate*64 + ct
    const int g    = nt >> 6;
    const int ct   = nt & 63;
    const int rsw  = lane ^ (lane >> 3);

    const v16bf* xb = reinterpret_cast<const v16bf*>(xp) + (((size_t)t * 8 + rt) * 16) * 32;
    const v16bf* wb = reinterpret_cast<const v16bf*>(Wkp);

    v8f acc = {};
    v16bf a_cur = xb[lane];
    v16bf b_cur = wb[((size_t)0 * 256 + nt) * 32 + rsw];
    #pragma unroll 4
    for (int ki = 0; ki < 16; ++ki) {
        v16bf a_n = a_cur, b_n = b_cur;
        if (ki < 15) {
            a_n = xb[(ki + 1) * 32 + lane];
            b_n = wb[((size_t)(ki + 1) * 256 + nt) * 32 + rsw];
        }
        acc = WMMA_BF16(a_cur, b_cur, acc);
        a_cur = a_n; b_cur = b_n;
    }
    v8f* dst = reinterpret_cast<v8f*>(zx) + ((((size_t)t * 8 + rt) * 64 + ct) * 4 + g) * 32 + lane;
    *dst = acc;
}

// ---------------- fused LSTM step (templated on K start) ----------------
// KSTART==0 : full GEMM ([x_t,h] @ Wk), zx unused.
// KSTART==16: h-only GEMM, accumulators seeded from precomputed z_x.
// grid = 64 blocks (ct), block = 256 threads = 8 waves (rt).
// B-tiles triple-buffered in LDS via global_load_async_to_lds_b128,
// prefetch distance 2, partial s_wait_asynccnt 0x1.
template<int KSTART>
__global__ __launch_bounds__(256) void lstm_step_t(
    const __bf16* __restrict__ xp, const __bf16* __restrict__ Wkp,
    const float* __restrict__ bk, const float* __restrict__ zx,
    const __bf16* __restrict__ hp_in, __bf16* __restrict__ hp_out,
    float* __restrict__ c_state, int t)
{
    __shared__ __align__(128) __bf16 smem[3 * 4 * 512];   // 3 x 4KB B-tile buffers

    constexpr int NK = 48 - KSTART;
    const int tid  = threadIdx.x;
    const int lane = tid & 31;
    const int rt   = tid >> 5;     // 0..7  row tile
    const int ct   = blockIdx.x;   // 0..63 col tile of H

    // --- async staging: thread tid copies the 16B chunk q of gate tile g ---
    const int g  = tid >> 6;                         // 0..3
    const int q  = tid & 63;                         // 16B chunk in (swizzled) tile
    const unsigned lds_base  = (unsigned)(uintptr_t)(&smem[0]);
    const unsigned lds_chunk = (unsigned)(g * 1024 + q * 16);
    const unsigned g_chunk0  = (unsigned)(((g * 64) + ct) * 1024 + q * 16);
    const unsigned long long wk_base = (unsigned long long)Wkp;

    const int rsw = lane ^ (lane >> 3);              // LDS read swizzle

    const v16bf* xbase = reinterpret_cast<const v16bf*>(xp) + (((size_t)t * 8 + rt) * 16) * 32;
    const v16bf* hbase = reinterpret_cast<const v16bf*>(hp_in) + ((size_t)rt * 32) * 32;
    auto loadA = [&](int ki) -> v16bf {
        const v16bf* ab = (ki < 16) ? (xbase + ki * 32) : (hbase + (ki - 16) * 32);
        return ab[lane];
    };
    auto stage = [&](int ki, int buf) {
        unsigned laddr = lds_base + (unsigned)(buf * 4096) + lds_chunk;
        unsigned goff  = g_chunk0 + (unsigned)ki * (256u * 1024u);
        asm volatile("global_load_async_to_lds_b128 %0, %1, %2"
                     :: "v"(laddr), "v"(goff), "s"(wk_base) : "memory");
    };

    // accumulators: zero, or seeded from precomputed z_x (fp32, acc layout)
    v8f acc0, acc1, acc2, acc3;
    if constexpr (KSTART == 16) {
        const v8f* zxt = reinterpret_cast<const v8f*>(zx) +
                         ((((size_t)t * 8 + rt) * 64 + ct) * 4) * 32;
        acc0 = zxt[0 * 32 + lane];
        acc1 = zxt[1 * 32 + lane];
        acc2 = zxt[2 * 32 + lane];
        acc3 = zxt[3 * 32 + lane];
    } else {
        acc0 = {}; acc1 = {}; acc2 = {}; acc3 = {};
    }

    // prologue: stage first two K-iters (prefetch depth 2)
    stage(KSTART + 0, 0);
    stage(KSTART + 1, 1);
    v16bf a_cur = loadA(KSTART);
    asm volatile("s_wait_asynccnt 0x1" ::: "memory");   // buffer 0 ready
    __syncthreads();

    #pragma unroll 3
    for (int kidx = 0; kidx < NK; ++kidx) {
        const int ki  = KSTART + kidx;
        const int buf = kidx % 3;

        if (kidx + 2 < NK) stage(ki + 2, (kidx + 2) % 3);

        v16bf a_nxt = a_cur;
        if (kidx < NK - 1) a_nxt = loadA(ki + 1);

        const v16bf* tb = reinterpret_cast<const v16bf*>(&smem[buf * 2048]);
        v16bf b0 = tb[0 * 32 + rsw];
        v16bf b1 = tb[1 * 32 + rsw];
        v16bf b2 = tb[2 * 32 + rsw];
        v16bf b3 = tb[3 * 32 + rsw];

        acc0 = WMMA_BF16(a_cur, b0, acc0);
        acc1 = WMMA_BF16(a_cur, b1, acc1);
        acc2 = WMMA_BF16(a_cur, b2, acc2);
        acc3 = WMMA_BF16(a_cur, b3, acc3);

        if (kidx < NK - 1) {
            if (kidx + 2 < NK) {
                // two staging ops in flight; the older (next iter's) must land
                asm volatile("s_wait_asynccnt 0x1" ::: "memory");
            } else {
                asm volatile("s_wait_asynccnt 0x0" ::: "memory");
            }
            __syncthreads();
        }
        a_cur = a_nxt;
    }

    // Gate math + state update.
    const int n  = lane & 15;
    const int mh = lane >> 4;
    const int col = ct * 16 + n;
    const float bi = bk[0 * LSTM_H + col];
    const float bj = bk[1 * LSTM_H + col];
    const float bf = bk[2 * LSTM_H + col];
    const float bo = bk[3 * LSTM_H + col];
    const int kt   = col >> 5;
    const int kk4  = col & 31;
    const int dl   = ((kk4 & 15) >= 8) ? 16 : 0;
    const int hidx = (kk4 & 7) + 8 * (kk4 >> 4);

    #pragma unroll
    for (int r = 0; r < 8; ++r) {
        const int m   = r + 8 * mh;
        const int row = rt * 16 + m;
        const float zi = acc0[r] + bi;
        const float zj = acc1[r] + bj;
        const float zf = acc2[r] + bf;
        const float zo = acc3[r] + bo;
        const size_t cidx = (size_t)row * LSTM_H + col;
        const float c_old = c_state[cidx];
        const float c_new = c_old * sigm(zf + 1.0f) + sigm(zi) * tanhf(zj);
        const float h_new = sigm(zo) * tanhf(c_new);
        c_state[cidx] = c_new;
        hp_out[((size_t)rt * 32 + kt) * 512 + (m + dl) * 16 + hidx] = (__bf16)h_new;
    }
}

// ---------------- final projection: out = hT @ w + b ----------------
__global__ __launch_bounds__(256) void proj_kernel(
    const __bf16* __restrict__ hp, const __bf16* __restrict__ wp,
    const float* __restrict__ bias, float* __restrict__ out)
{
    const int lane = threadIdx.x & 31;
    const int rt   = threadIdx.x >> 5;
    const int ct   = blockIdx.x;           // 0..62
    const int rsw  = lane ^ (lane >> 3);

    v8f acc = {};
    const v16bf* hpt = reinterpret_cast<const v16bf*>(hp);
    const v16bf* wpt = reinterpret_cast<const v16bf*>(wp);

    v16bf a_cur = hpt[((size_t)rt * 32 + 0) * 32 + lane];
    v16bf b_cur = wpt[((size_t)0 * 63 + ct) * 32 + rsw];
    #pragma unroll 4
    for (int ki = 0; ki < 32; ++ki) {
        v16bf a_nxt = a_cur, b_nxt = b_cur;
        if (ki < 31) {
            a_nxt = hpt[((size_t)rt * 32 + (ki + 1)) * 32 + lane];
            b_nxt = wpt[((size_t)(ki + 1) * 63 + ct) * 32 + rsw];
        }
        acc = WMMA_BF16(a_cur, b_cur, acc);
        a_cur = a_nxt;
        b_cur = b_nxt;
    }

    const int n  = lane & 15;
    const int mh = lane >> 4;
    const int col = ct * 16 + n;
    if (col < LSTM_C) {
        const float bb = bias[col];
        #pragma unroll
        for (int r = 0; r < 8; ++r) {
            const int row = rt * 16 + r + 8 * mh;
            out[(size_t)row * LSTM_C + col] = acc[r] + bb;
        }
    }
}

extern "C" void kernel_launch(void* const* d_in, const int* in_sizes, int n_in,
                              void* d_out, int out_size, void* d_ws, size_t ws_size,
                              hipStream_t stream) {
    const float* x  = (const float*)d_in[0];   // [128,256,512]
    const float* Wk = (const float*)d_in[1];   // [1536,4096]
    const float* bk = (const float*)d_in[2];   // [4096]
    const float* w  = (const float*)d_in[3];   // [1024,1000]
    const float* b  = (const float*)d_in[4];   // [1000]
    float* out = (float*)d_out;                // [128,1000]

    char* ws = (char*)d_ws;
    __bf16* xp  = (__bf16*)(ws + WS_XP);
    __bf16* Wkp = (__bf16*)(ws + WS_WKP);
    __bf16* wp  = (__bf16*)(ws + WS_WP);
    __bf16* hp0 = (__bf16*)(ws + WS_HP0);
    __bf16* hp1 = (__bf16*)(ws + WS_HP1);
    float*  cst = (float*)(ws + WS_C);
    float*  zxp = (float*)(ws + WS_ZX);

    // enough workspace to hoist the x-part out of the recurrence?
    const bool pre = (ws_size >= (size_t)WS_ZX_END);

    // zero h0/h1/c (contiguous region, capture-safe async memset)
    hipMemsetAsync(ws + WS_HP0, 0, WS_ZX - WS_HP0, stream);

    // pack inputs
    pack_x_kernel <<<(LSTM_B * LSTM_T * LSTM_D) / 256, 256, 0, stream>>>(x, xp);
    pack_wk_kernel<<<((LSTM_D + LSTM_H) * 4 * LSTM_H) / 256, 256, 0, stream>>>(Wk, Wkp);
    pack_w_kernel <<<(LSTM_H * LSTM_CP + 255) / 256, 256, 0, stream>>>(w, wp);

    if (pre) {
        // parallel x-part GEMM for all timesteps: z_x = x @ Wx
        zx_kernel<<<LSTM_T * 256, 256, 0, stream>>>(xp, Wkp, zxp);
        for (int t = 0; t < LSTM_T; ++t) {
            const __bf16* hin  = (t & 1) ? hp1 : hp0;
            __bf16*       hout = (t & 1) ? hp0 : hp1;
            lstm_step_t<16><<<LSTM_H / 16, 256, 0, stream>>>(xp, Wkp, bk, zxp, hin, hout, cst, t);
        }
    } else {
        for (int t = 0; t < LSTM_T; ++t) {
            const __bf16* hin  = (t & 1) ? hp1 : hp0;
            __bf16*       hout = (t & 1) ? hp0 : hp1;
            lstm_step_t<0><<<LSTM_H / 16, 256, 0, stream>>>(xp, Wkp, bk, nullptr, hin, hout, cst, t);
        }
    }

    // t=255 (odd) wrote hp0 -> final h lives in hp0
    proj_kernel<<<LSTM_CP / 16, 256, 0, stream>>>(hp0, wp, b, out);
}
